// StackedLSTMSequenceModel_24232205484321
// MI455X (gfx1250) — matrix operand order repaired and checked by reading, not compiled, not run on hardware
//
#include <hip/hip_runtime.h>
#include <hip/hip_bf16.h>

// ---------------------------------------------------------------------------
// Stacked LSTM (B=512, T=1024, IN=64, H=512, OUT=2, 2 layers, 5 future steps)
// Persistent workgroup per 16-row batch tile. fp16 WMMA
// (v_wmma_f32_16x16x32_f16) with fp32 accumulation. Weights pre-transposed to
// fp16 in d_ws and streamed from L2 as B fragments (global_load_b128, 4-deep
// software pipeline, per-tile constants folded into the instruction offset);
// hidden state fp16 in ping-pong LDS buffers (A operand); cell state in
// wave-private LDS (frees 64 VGPRs for load pipelining); gate biases folded
// into the accumulator init.
// ---------------------------------------------------------------------------

typedef _Float16 half8  __attribute__((ext_vector_type(8)));
typedef _Float16 half16 __attribute__((ext_vector_type(16)));
typedef float    float8 __attribute__((ext_vector_type(8)));

#define B_   512
#define T_   1024
#define IN_  64
#define H_   512
#define G4H_ 2048   // 4 gates * H
#define OUT_ 2
#define BT_  16     // batch tile rows (WMMA M)
#define NW_  8      // waves per workgroup (wave32)

__device__ __forceinline__ float sigm(float x) { return 1.0f / (1.0f + __expf(-x)); }
__device__ __forceinline__ float tanh_fast(float x) {
    // tanh(x) = 1 - 2/(exp(2x)+1); safe for large |x| (exp->inf or 0)
    float e = __expf(2.0f * x);
    return 1.0f - 2.0f / (e + 1.0f);
}

__device__ __forceinline__ half16 cat8(half8 lo, half8 hi) {
    return __builtin_shufflevector(lo, hi, 0,1,2,3,4,5,6,7,8,9,10,11,12,13,14,15);
}

// A fragment (16x32 fp16, M x K), row-major source with leading dim ld.
// Layout: lanes 0-15 hold M=lane, K = kbase+{0..15};
//         lanes 16-31 hold M=lane-16, K = kbase+{16..31}.
__device__ __forceinline__ half16 load_a(const _Float16* base, int ld, int kbase, int lane) {
    int m  = lane & 15;
    int k0 = kbase + ((lane & 16) ? 16 : 0);
    const _Float16* p = base + m * ld + k0;
    half8 lo = *(const half8*)(p);        // K = k0 .. k0+7
    half8 hi = *(const half8*)(p + 8);    // K = k0+8 .. k0+15
    return cat8(lo, hi);
}

// B fragment: 16 contiguous fp16 at bp + eoff (eoff compile-time per tile so
// it folds into the global_load 24-bit instruction offset).
__device__ __forceinline__ half16 load_b16(const _Float16* bp, int eoff) {
    half8 lo = *(const half8*)(bp + eoff);
    half8 hi = *(const half8*)(bp + eoff + 8);
    return cat8(lo, hi);
}

__device__ __forceinline__ float8 wmma_f16(half16 a, half16 b, float8 c) {
    // D = A*B + C : v_wmma_f32_16x16x32_f16
    return __builtin_amdgcn_wmma_f32_16x16x32_f16(false, a, false, b, (short)0, c, false, false);
}

// Element offset of gate-tile tt inside W^T (columns of 4 concatenated gates).
__device__ __forceinline__ constexpr int tile_eoff(int tt, int K) {
    return ((tt >> 2) * 512 + (tt & 3) * 16) * K;
}

// Gate-tile GEMM for one wave's 64 hidden columns (16 tiles = 4 gates x 4):
//   acc = bias_splat + x_tile(16 x xK) @ Wx + h_tile(16 x 512) @ Wh
// B fragments in a 4-deep software pipeline: load(tt+4) issues before wmma(tt).
__device__ __forceinline__ void gemm_gates(float8 acc[16],
                                           const _Float16* xa, int xK, const _Float16* wx,
                                           const _Float16* ha, const _Float16* wh,
                                           const float* sh_bias, int j0, int lane, int zoff) {
    const int nn = lane & 15;

    // Init accumulators with the gate bias (same N column for all 8 rows).
    #pragma unroll
    for (int tt = 0; tt < 16; ++tt) {
        float b = sh_bias[((tt >> 2) << 9) + j0 + ((tt & 3) << 4) + nn];
        float8 v;
        #pragma unroll
        for (int e = 0; e < 8; ++e) v[e] = b;
        acc[tt] = v;
    }

    // x-part (K = xK), then h-part (K = 512); same inner structure.
    #pragma unroll 1
    for (int kc = 0; kc < xK; kc += 32) {
        half16 a = load_a(xa, xK, kc, lane);
        const _Float16* bp = wx + (j0 + nn) * xK + kc + ((lane & 16) ? 16 : 0) + zoff;
        half16 bq[4];
        #pragma unroll
        for (int i = 0; i < 4; ++i) bq[i] = load_b16(bp, tile_eoff(i, xK));
        #pragma unroll
        for (int tt = 0; tt < 16; ++tt) {
            half16 bnew = bq[tt & 3];
            if (tt + 4 < 16) bnew = load_b16(bp, tile_eoff(tt + 4, xK));
            acc[tt] = wmma_f16(a, bq[tt & 3], acc[tt]);
            bq[tt & 3] = bnew;
        }
    }
    #pragma unroll 1
    for (int kc = 0; kc < H_; kc += 32) {
        half16 a = load_a(ha, H_, kc, lane);
        const _Float16* bp = wh + (j0 + nn) * H_ + kc + ((lane & 16) ? 16 : 0) + zoff;
        half16 bq[4];
        #pragma unroll
        for (int i = 0; i < 4; ++i) bq[i] = load_b16(bp, tile_eoff(i, H_));
        #pragma unroll
        for (int tt = 0; tt < 16; ++tt) {
            half16 bnew = bq[tt & 3];
            if (tt + 4 < 16) bnew = load_b16(bp, tile_eoff(tt + 4, H_));
            acc[tt] = wmma_f16(a, bq[tt & 3], acc[tt]);
            bq[tt & 3] = bnew;
        }
    }
}

// LSTM gate math + immediate fp16 store of h into the ping-pong target.
// acc tiles: [0..3]=i, [4..7]=f, [8..11]=o, [12..15]=g (bias already in acc).
// Cell state lives in wave-private LDS (cw[jj][lane]); DS ops are in-order
// within a wave, so no barriers are needed around it.
template <bool SAVE>
__device__ __forceinline__ void cell_update(const float8 acc[16], float8 (*cw)[32],
                                            _Float16* hdst, int j0, int lane,
                                            float8 hsave[4]) {
    const int half = (lane >> 4) << 3;
    const int nn   = lane & 15;
    #pragma unroll
    for (int jj = 0; jj < 4; ++jj) {
        float8 cv = cw[jj][lane];
        float8 hv8;
        #pragma unroll
        for (int e = 0; e < 8; ++e) {
            float iv = sigm(acc[jj][e]);
            float fv = sigm(acc[4 + jj][e]);
            float ov = sigm(acc[8 + jj][e]);
            float gv = tanh_fast(acc[12 + jj][e]);
            float cn = fv * cv[e] + iv * gv;
            cv[e]  = cn;
            hv8[e] = ov * tanh_fast(cn);
        }
        cw[jj][lane] = cv;
        if (SAVE) hsave[jj] = hv8;
        #pragma unroll
        for (int e = 0; e < 8; ++e) {
            hdst[(size_t)(e + half) * H_ + j0 + (jj << 4) + nn] = (_Float16)hv8[e];
        }
    }
}

// ------------------------------ main kernel --------------------------------
__global__ __launch_bounds__(256, 1)
void lstm_persistent(const float* __restrict__ in_seq,
                     const _Float16* __restrict__ wx0, const _Float16* __restrict__ wh0,
                     const _Float16* __restrict__ wx1, const _Float16* __restrict__ wh1,
                     const float* __restrict__ bias0, const float* __restrict__ bias1,
                     const float* __restrict__ wout,  const float* __restrict__ bout,
                     const int* __restrict__ fsp,     float* __restrict__ dout) {
    const int tid     = threadIdx.x;
    const int lane    = tid & 31;
    const int w       = tid >> 5;
    const int j0      = w << 6;              // this wave's 64 hidden columns
    const int rowbase = blockIdx.x * BT_;

    __shared__ _Float16 sh_x[BT_][IN_];
    __shared__ _Float16 sh_h0[2][BT_][H_];   // ping-pong
    __shared__ _Float16 sh_h1[2][BT_][H_];   // ping-pong
    __shared__ float    sh_bias[2][G4H_];
    __shared__ float8   sh_c0[NW_][4][32];   // wave-private cell state, layer 0
    __shared__ float8   sh_c1[NW_][4][32];   // wave-private cell state, layer 1
    __shared__ float    sh_wpart[NW_][BT_][OUT_];

    for (int i = tid; i < BT_ * H_; i += 256) {
        (&sh_h0[0][0][0])[i] = (_Float16)0.0f;   // only parity 0 needs init
        (&sh_h1[0][0][0])[i] = (_Float16)0.0f;
    }
    for (int i = tid; i < G4H_; i += 256) {
        sh_bias[0][i] = bias0[i];
        sh_bias[1][i] = bias1[i];
    }
    {
        const float8 z8 = {0.f,0.f,0.f,0.f,0.f,0.f,0.f,0.f};
        #pragma unroll
        for (int jj = 0; jj < 4; ++jj) {
            sh_c0[w][jj][lane] = z8;
            sh_c1[w][jj][lane] = z8;
        }
    }

    int fut = fsp[0];
    if (fut < 0) fut = 0;
    if (fut > 8) fut = 8;

    for (int t = 0; t < T_ + fut; ++t) {
        // Opaque zero offset: keeps weight addresses loop-variant (no LICM
        // hoist -> stream from L2) while preserving kernarg provenance so the
        // loads stay global_load_b128, not flat.
        int zoff = 0;
        asm volatile("" : "+v"(zoff));

        const int  p      = t & 1;
        const bool future = (t >= T_);
        if (!future) {
            // Stage x_t tile (16 rows x 64) as fp16 into LDS.
            int r  = tid >> 4;
            int cb = (tid & 15) << 2;
            const float4 xv = *(const float4*)(in_seq + ((size_t)(rowbase + r) * T_ + t) * IN_ + cb);
            sh_x[r][cb + 0] = (_Float16)xv.x;
            sh_x[r][cb + 1] = (_Float16)xv.y;
            sh_x[r][cb + 2] = (_Float16)xv.z;
            sh_x[r][cb + 3] = (_Float16)xv.w;
        }
        __syncthreads();                                   // x + h(t-1) visible

        float8 acc[16];
        float8 hval[4];

        // ---- layer 0: read h0[p], write h0[1-p] ----
        gemm_gates(acc, &sh_x[0][0], IN_, wx0, &sh_h0[p][0][0], wh0,
                   &sh_bias[0][0], j0, lane, zoff);
        cell_update<false>(acc, sh_c0[w], &sh_h0[1 - p][0][0], j0, lane, hval);
        __syncthreads();                                   // h0(t) ready

        // ---- layer 1: x-input = h0[1-p] (new), read h1[p], write h1[1-p] ----
        gemm_gates(acc, &sh_h0[1 - p][0][0], H_, wx1, &sh_h1[p][0][0], wh1,
                   &sh_bias[1][0], j0, lane, zoff);
        if (future) {
            cell_update<true>(acc, sh_c1[w], &sh_h1[1 - p][0][0], j0, lane, hval);

            // Output-projection partials from fp32 h (before fp16 rounding).
            float pr[OUT_][8];
            #pragma unroll
            for (int o = 0; o < OUT_; ++o)
                #pragma unroll
                for (int e = 0; e < 8; ++e) pr[o][e] = 0.0f;
            #pragma unroll
            for (int jj = 0; jj < 4; ++jj) {
                int col  = j0 + (jj << 4) + (lane & 15);
                float w0 = wout[col + zoff];
                float w1 = wout[H_ + col + zoff];
                #pragma unroll
                for (int e = 0; e < 8; ++e) {
                    pr[0][e] += hval[jj][e] * w0;
                    pr[1][e] += hval[jj][e] * w1;
                }
            }
            #pragma unroll
            for (int m = 1; m < 16; m <<= 1) {             // butterfly over 16 lanes
                #pragma unroll
                for (int o = 0; o < OUT_; ++o)
                    #pragma unroll
                    for (int e = 0; e < 8; ++e)
                        pr[o][e] += __shfl_xor(pr[o][e], m);
            }
            if ((lane & 15) == 0) {
                int half = (lane >> 4) << 3;
                #pragma unroll
                for (int e = 0; e < 8; ++e) {
                    sh_wpart[w][half + e][0] = pr[0][e];
                    sh_wpart[w][half + e][1] = pr[1][e];
                }
            }
        } else {
            cell_update<false>(acc, sh_c1[w], &sh_h1[1 - p][0][0], j0, lane, hval);
        }
        __syncthreads();                                   // h1(t) + partials ready

        if (future && tid < BT_ * OUT_) {
            int r = tid >> 1, o = tid & 1;
            float s = bout[o];
            #pragma unroll
            for (int wv = 0; wv < NW_; ++wv) s += sh_wpart[wv][r][o];  // fixed order
            int st = t - T_;
            dout[((size_t)(rowbase + r) * fut + st) * OUT_ + o] = s;
            sh_x[r][o] = (_Float16)s;   // x_{next} = [out, tail]; tail kept in LDS
        }
    }
}

// --------------------------- weight prep kernels ---------------------------
// dst[n*K + k] = (fp16) src[k*512 + n]   (src is K x 512 row-major)
__global__ void k_transpose_f16(const float* __restrict__ src, _Float16* __restrict__ dst, int K) {
    int idx = blockIdx.x * 256 + threadIdx.x;
    if (idx >= K * 512) return;
    int n = idx & 511;
    int k = idx >> 9;
    dst[(size_t)n * K + k] = (_Float16)src[(size_t)k * 512 + n];
}

__global__ void k_copy_bias(const float* __restrict__ src, float* __restrict__ dst) {
    int i = blockIdx.x * 256 + threadIdx.x;
    if (i < 512) dst[i] = src[i];
}

// -------------------------------- launcher ---------------------------------
extern "C" void kernel_launch(void* const* d_in, const int* in_sizes, int n_in,
                              void* d_out, int out_size, void* d_ws, size_t ws_size,
                              hipStream_t stream) {
    // d_in layout (setup_inputs dict order, params dicts flattened in key order):
    // 0:input_seq | 1..12: params0 {Wii,Whi,bi,Wif,Whf,bf,Wio,Who,bo,Wig,Whg,bg}
    // 13..24: params1 same | 25:W_out | 26:b_out | 27:future_steps
    const float* in_seq = (const float*)d_in[0];

    // Workspace: fp16 transposed weight blocks + fp32 concatenated biases
    const size_t need = ((size_t)G4H_ * IN_ + 3u * (size_t)G4H_ * H_) * sizeof(_Float16)
                      + 2u * (size_t)G4H_ * sizeof(float);
    if (ws_size < need) return;

    _Float16* wx0 = (_Float16*)d_ws;                    // 2048 x 64
    _Float16* wh0 = wx0 + (size_t)G4H_ * IN_;           // 2048 x 512
    _Float16* wx1 = wh0 + (size_t)G4H_ * H_;            // 2048 x 512
    _Float16* wh1 = wx1 + (size_t)G4H_ * H_;            // 2048 x 512
    float* bias0  = (float*)(wh1 + (size_t)G4H_ * H_);  // 2048
    float* bias1  = bias0 + G4H_;                       // 2048

    const int wxi0[4] = {1, 4, 7, 10},  whi0[4] = {2, 5, 8, 11},  bi0[4] = {3, 6, 9, 12};
    const int wxi1[4] = {13, 16, 19, 22}, whi1[4] = {14, 17, 20, 23}, bi1[4] = {15, 18, 21, 24};

    const int gx0 = (IN_ * 512 + 255) / 256;   // K=64 transposes
    const int gh  = (H_ * 512 + 255) / 256;    // K=512 transposes
    for (int g = 0; g < 4; ++g) {
        k_transpose_f16<<<gx0, 256, 0, stream>>>((const float*)d_in[wxi0[g]],
                                                 wx0 + (size_t)g * 512 * IN_, IN_);
        k_transpose_f16<<<gh, 256, 0, stream>>>((const float*)d_in[whi0[g]],
                                                wh0 + (size_t)g * 512 * H_, H_);
        k_transpose_f16<<<gh, 256, 0, stream>>>((const float*)d_in[wxi1[g]],
                                                wx1 + (size_t)g * 512 * H_, H_);
        k_transpose_f16<<<gh, 256, 0, stream>>>((const float*)d_in[whi1[g]],
                                                wh1 + (size_t)g * 512 * H_, H_);
        k_copy_bias<<<2, 256, 0, stream>>>((const float*)d_in[bi0[g]], bias0 + g * 512);
        k_copy_bias<<<2, 256, 0, stream>>>((const float*)d_in[bi1[g]], bias1 + g * 512);
    }

    lstm_persistent<<<B_ / BT_, 256, 0, stream>>>(
        in_seq, wx0, wh0, wx1, wh1, bias0, bias1,
        (const float*)d_in[25], (const float*)d_in[26],
        (const int*)d_in[27], (float*)d_out);
}
